// VQ2_21586505630025
// MI455X (gfx1250) — matrix operand order repaired and checked by reading, not compiled, not run on hardware
//
#include <hip/hip_runtime.h>
#include <hip/hip_bf16.h>

// ---------------------------------------------------------------------------
// VQ-VAE forward for MI455X (gfx1250, wave32, WMMA).
#define BATCH 16384
#define D_IN  512
#define HID   64
#define CDIM  256
#define KPROT 2048

typedef __attribute__((ext_vector_type(16))) _Float16 v16h;
typedef __attribute__((ext_vector_type(8)))  float    v8f;

// ---------------------------------------------------------------------------
// Fragment helpers for the CDNA5 wave32 WMMA layouts (CDNA5 ISA
// §7.12.2), written so each lane's data is fetched as 16-byte chunks.

// A-matrix 16x32 f16 from LDS/row-major buffer with leading dim ld (halfs).
// Lane layout: elements 0..7  <- K = k0 + 8*hi + (0..7)
//              elements 8..15 <- K = k0 + 16 + 8*hi + (0..7)
// Both runs are contiguous -> two b128 loads per lane.
__device__ inline v16h load_a_frag(const _Float16* base, int ld, int k0) {
  const int lane = threadIdx.x & 31;
  const int m    = lane & 15;
  const int hi   = lane >> 4;
  const _Float16* p = base + m * ld + k0 + hi * 8;
  union { uint4 q[2]; v16h v; } u;
  u.q[0] = *reinterpret_cast<const uint4*>(p);
  u.q[1] = *reinterpret_cast<const uint4*>(p + 16);
  return u.v;
}

// B-matrix 32x16 f16 from an N-major source T[N][Kdim] (ldk = Kdim in halfs):
// B[k,n] = T[n0+n][k0+k]. Lane layout: elements 0..7 <- K = k0 + 8*hi + (0..7),
// elements 8..15 <- K = k0 + 16 + 8*hi + (0..7) -> two b128 loads per lane.
__device__ inline v16h load_b_frag_T(const _Float16* __restrict__ T, int ldk,
                                     int k0, int n0) {
  const int lane = threadIdx.x & 31;
  const int n    = lane & 15;
  const int hi   = lane >> 4;
  const _Float16* p = T + (size_t)(n0 + n) * ldk + k0 + hi * 8;
  union { uint4 q[2]; v16h v; } u;
  u.q[0] = *reinterpret_cast<const uint4*>(p);
  u.q[1] = *reinterpret_cast<const uint4*>(p + 16);
  return u.v;
}

// C/D 16x16 f32: VGPR r -> row = r + 8*hi, col = n0 + (lane&15).
__device__ inline void store_d_lds(_Float16* dst, int ld, int n0, v8f a,
                                   const float* __restrict__ bias, bool relu) {
  const int lane = threadIdx.x & 31;
  const int n    = lane & 15;
  const int hi   = lane >> 4;
  const float bv = bias[n0 + n];
#pragma unroll
  for (int r = 0; r < 8; ++r) {
    const int row = r + hi * 8;
    float v = a[r] + bv;
    if (relu && v < 0.f) v = 0.f;
    dst[row * ld + n0 + n] = (_Float16)v;
  }
}

__device__ inline v8f wmma_f16(v16h a, v16h b, v8f c) {
  return __builtin_amdgcn_wmma_f32_16x16x32_f16(false, a, false, b, (short)0, c,
                                                false, false);
}

// ---------------------------------------------------------------------------
// k_prep: f32->f16 conversion with TRANSPOSE of all weights to [N][K] so that
// every WMMA B-fragment load is a contiguous b128 pair; proto norms; zeroing.
__global__ void k_prep(const float* __restrict__ Wemb, const float* __restrict__ W0,
                       const float* __restrict__ W1, const float* __restrict__ Wmu,
                       const float* __restrict__ protos,
                       _Float16* tWemb,   // [HID][D_IN]
                       _Float16* tW0,     // [HID][HID]
                       _Float16* tW1,     // [CDIM][HID]
                       _Float16* tWmu,    // [CDIM][CDIM]
                       _Float16* oP,      // [KPROT][CDIM] (protos, no transpose)
                       float* pnorm, float* S, float* Tnd, float* lse_sum) {
  const int tid  = blockIdx.x * blockDim.x + threadIdx.x;
  const int nthr = gridDim.x * blockDim.x;
  for (int i = tid; i < D_IN * HID; i += nthr) {
    const int k = i / HID, n = i % HID;
    tWemb[(size_t)n * D_IN + k] = (_Float16)Wemb[i];
  }
  for (int i = tid; i < HID * HID; i += nthr) {
    const int k = i / HID, n = i % HID;
    tW0[(size_t)n * HID + k] = (_Float16)W0[i];
  }
  for (int i = tid; i < HID * CDIM; i += nthr) {
    const int k = i / CDIM, n = i % CDIM;
    tW1[(size_t)n * HID + k] = (_Float16)W1[i];
  }
  for (int i = tid; i < CDIM * CDIM; i += nthr) {
    const int k = i / CDIM, n = i % CDIM;
    tWmu[(size_t)n * CDIM + k] = (_Float16)Wmu[i];
  }
  for (int i = tid; i < KPROT * CDIM; i += nthr) oP[i] = (_Float16)protos[i];
  for (int k = tid; k < KPROT; k += nthr) {
    const float* p = protos + (size_t)k * CDIM;
    float s = 0.f;
    for (int c = 0; c < CDIM; ++c) s += p[c] * p[c];
    pnorm[k] = s;
    S[k] = 0.f;
    Tnd[k] = 0.f;
  }
  if (tid == 0) *lse_sum = 0.f;
}

// ---------------------------------------------------------------------------
// k_enc: fused 4-layer MLP, one wave per 16-row tile, all GEMMs via WMMA.
__global__ __launch_bounds__(32) void k_enc(
    const float* __restrict__ x,
    const _Float16* __restrict__ tWemb, const float* __restrict__ bemb,
    const _Float16* __restrict__ tW0, const float* __restrict__ b0,
    const _Float16* __restrict__ tW1, const float* __restrict__ b1,
    const _Float16* __restrict__ tWmu, const float* __restrict__ bmu,
    _Float16* __restrict__ mu_h, float* __restrict__ mu_norm) {
  __shared__ __align__(16) _Float16 xt[16 * D_IN];
  __shared__ __align__(16) _Float16 h1[16 * HID];
  __shared__ __align__(16) _Float16 h2[16 * HID];
  __shared__ __align__(16) _Float16 h3[16 * CDIM];
  __shared__ float rn[16];

  const int b0r  = blockIdx.x * 16;
  const int lane = threadIdx.x;
  const int n    = lane & 15;
  const int hi   = lane >> 4;

  // stage x tile, f32 -> f16, vectorized: float4 in, 4 packed halfs out.
  for (int i = lane * 4; i < 16 * D_IN; i += 32 * 4) {
    const float4 f = *reinterpret_cast<const float4*>(x + (size_t)b0r * D_IN + i);
    union { _Float16 h[4]; uint2 d; } u;
    u.h[0] = (_Float16)f.x; u.h[1] = (_Float16)f.y;
    u.h[2] = (_Float16)f.z; u.h[3] = (_Float16)f.w;
    *reinterpret_cast<uint2*>(xt + i) = u.d;
  }
  if (lane < 16) rn[lane] = 0.f;
  __syncthreads();

  // h1 = x @ W_emb + b_emb   (no ReLU on first layer)
#pragma unroll
  for (int t = 0; t < 4; ++t) {
    v8f acc = {0, 0, 0, 0, 0, 0, 0, 0};
#pragma unroll
    for (int kk = 0; kk < 16; ++kk)
      acc = wmma_f16(load_a_frag(xt, D_IN, kk * 32),
                     load_b_frag_T(tWemb, D_IN, kk * 32, t * 16), acc);
    store_d_lds(h1, HID, t * 16, acc, bemb, false);
  }
  __syncthreads();

  // h2 = relu(h1 @ W0 + b0)
#pragma unroll
  for (int t = 0; t < 4; ++t) {
    v8f acc = {0, 0, 0, 0, 0, 0, 0, 0};
#pragma unroll
    for (int kk = 0; kk < 2; ++kk)
      acc = wmma_f16(load_a_frag(h1, HID, kk * 32),
                     load_b_frag_T(tW0, HID, kk * 32, t * 16), acc);
    store_d_lds(h2, HID, t * 16, acc, b0, true);
  }
  __syncthreads();

  // h3 = relu(h2 @ W1 + b1)
#pragma unroll
  for (int t = 0; t < 16; ++t) {
    v8f acc = {0, 0, 0, 0, 0, 0, 0, 0};
#pragma unroll
    for (int kk = 0; kk < 2; ++kk)
      acc = wmma_f16(load_a_frag(h2, HID, kk * 32),
                     load_b_frag_T(tW1, HID, kk * 32, t * 16), acc);
    store_d_lds(h3, CDIM, t * 16, acc, b1, true);
  }
  __syncthreads();

  // mu = h3 @ W_mu + b_mu ; store f16 mu and accumulate ||mu_b||^2
#pragma unroll
  for (int t = 0; t < 16; ++t) {
    v8f acc = {0, 0, 0, 0, 0, 0, 0, 0};
#pragma unroll
    for (int kk = 0; kk < 8; ++kk)
      acc = wmma_f16(load_a_frag(h3, CDIM, kk * 32),
                     load_b_frag_T(tWmu, CDIM, kk * 32, t * 16), acc);
    const float bv = bmu[t * 16 + n];
#pragma unroll
    for (int r = 0; r < 8; ++r) {
      const int row = r + hi * 8;
      const float v = acc[r] + bv;
      mu_h[(size_t)(b0r + row) * CDIM + t * 16 + n] = (_Float16)v;
      atomicAdd(&rn[row], v * v);   // ds_add_f32
    }
  }
  __syncthreads();
  if (lane < 16) mu_norm[b0r + lane] = rn[lane];
}

// ---------------------------------------------------------------------------
// k_dist: distance GEMM + softmax stats + argmax gather. 8 waves x 256 cols,
// 16 rows per block; neg_d tile register-resident (acc[16] = 128 VGPRs/lane).
__global__ __launch_bounds__(256) void k_dist(
    const _Float16* __restrict__ mu_h, const float* __restrict__ mu_norm,
    const _Float16* __restrict__ ph, const float* __restrict__ pnorm,
    const float* __restrict__ gumbel, const float* __restrict__ protos_f32,
    float* __restrict__ S, float* __restrict__ Tnd,
    float* __restrict__ lse_sum, float* __restrict__ out_q) {
  __shared__ __align__(16) _Float16 mu_t[16 * CDIM];
  __shared__ float mn[16];
  __shared__ float redf[8][16];
  __shared__ float redv[8][16];
  __shared__ int   redi[8][16];
  __shared__ float rowmax_s[16];
  __shared__ float lse_s[16];
  __shared__ int   aidx_s[16];

  const int b0   = blockIdx.x * 16;
  const int tid  = threadIdx.x;
  const int lane = tid & 31;
  const int w    = tid >> 5;
  const int nloc = lane & 15;
  const int hi   = lane >> 4;
  const int colbase = w * 256;

  // stage mu tile (b128 copies)
  {
    const uint4* src = reinterpret_cast<const uint4*>(mu_h + (size_t)b0 * CDIM);
    uint4* dst = reinterpret_cast<uint4*>(mu_t);
    for (int i = tid; i < 16 * CDIM / 8; i += 256) dst[i] = src[i];
  }
  if (tid < 16) mn[tid] = mu_norm[b0 + tid];
  __syncthreads();

  v8f acc[16];
  float rmax[8], av[8];
  int ai[8];
#pragma unroll
  for (int r = 0; r < 8; ++r) { rmax[r] = -3.4e38f; av[r] = -3.4e38f; ai[r] = 0; }

  // ---- sweep 1: WMMA GEMM -> neg_d (in regs); rowmax, argmax(nd+gumbel), T_k
#pragma unroll
  for (int t = 0; t < 16; ++t) {
    const int n0  = colbase + t * 16;
    const int col = n0 + nloc;
    if (t + 1 < 16)
      __builtin_prefetch(ph + (size_t)(n0 + 16 + nloc) * CDIM, 0, 1);
    v8f a = {0, 0, 0, 0, 0, 0, 0, 0};
#pragma unroll
    for (int kk = 0; kk < 8; ++kk)
      a = wmma_f16(load_a_frag(mu_t, CDIM, kk * 32),
                   load_b_frag_T(ph, CDIM, kk * 32, n0), a);
    const float pn = pnorm[col];
    float cs = 0.f;
#pragma unroll
    for (int r = 0; r < 8; ++r) {
      const int row = r + hi * 8;
      const float nd = 2.f * a[r] - mn[row] - pn;
      a[r] = nd;
      rmax[r] = fmaxf(rmax[r], nd);
      const float v = nd + gumbel[(size_t)(b0 + row) * KPROT + col];
      if (v > av[r]) { av[r] = v; ai[r] = col; }
      cs += nd;
    }
    acc[t] = a;
    const float other = __shfl_xor(cs, 16, 32);
    if (hi == 0) atomicAdd(&Tnd[col], cs + other);
  }

  // cross-lane reduce (within 16-lane halves; halves hold disjoint row sets)
#pragma unroll
  for (int r = 0; r < 8; ++r) {
    float v = rmax[r];
    for (int m = 1; m < 16; m <<= 1) v = fmaxf(v, __shfl_xor(v, m, 32));
    rmax[r] = v;
    float va = av[r]; int ia = ai[r];
    for (int m = 1; m < 16; m <<= 1) {
      const float ov = __shfl_xor(va, m, 32);
      const int   oi = __shfl_xor(ia, m, 32);
      if (ov > va) { va = ov; ia = oi; }
    }
    av[r] = va; ai[r] = ia;
  }
  if (lane == 0)
    for (int r = 0; r < 8; ++r) { redf[w][r] = rmax[r]; redv[w][r] = av[r]; redi[w][r] = ai[r]; }
  if (lane == 16)
    for (int r = 0; r < 8; ++r) { redf[w][8 + r] = rmax[r]; redv[w][8 + r] = av[r]; redi[w][8 + r] = ai[r]; }
  __syncthreads();
  if (tid < 16) {
    float v = redf[0][tid], va = redv[0][tid];
    int ia = redi[0][tid];
    for (int ww = 1; ww < 8; ++ww) {
      v = fmaxf(v, redf[ww][tid]);
      if (redv[ww][tid] > va) { va = redv[ww][tid]; ia = redi[ww][tid]; }
    }
    rowmax_s[tid] = v;
    aidx_s[tid] = ia;
  }
  __syncthreads();

  // ---- sweep 2: row sum-exp -> lse (no GEMM recompute: acc is in regs)
  float sume[8];
#pragma unroll
  for (int r = 0; r < 8; ++r) sume[r] = 0.f;
#pragma unroll
  for (int t = 0; t < 16; ++t) {
#pragma unroll
    for (int r = 0; r < 8; ++r) {
      const int row = r + hi * 8;
      sume[r] += __expf(acc[t][r] - rowmax_s[row]);
    }
  }
#pragma unroll
  for (int r = 0; r < 8; ++r) {
    float v = sume[r];
    for (int m = 1; m < 16; m <<= 1) v += __shfl_xor(v, m, 32);
    sume[r] = v;
  }
  if (lane == 0)  for (int r = 0; r < 8; ++r) redf[w][r] = sume[r];
  if (lane == 16) for (int r = 0; r < 8; ++r) redf[w][8 + r] = sume[r];
  __syncthreads();
  if (tid < 16) {
    float se = 0.f;
    for (int ww = 0; ww < 8; ++ww) se += redf[ww][tid];
    lse_s[tid] = rowmax_s[tid] + __logf(se);
  }
  __syncthreads();
  if (tid == 0) {
    float t = 0.f;
    for (int r = 0; r < 16; ++r) t += lse_s[r];
    atomicAdd(lse_sum, t);   // global_atomic_add_f32
  }

  // ---- sweep 3: S_k += sum_rows softmax = exp(nd - lse)
#pragma unroll
  for (int t = 0; t < 16; ++t) {
    const int col = colbase + t * 16 + nloc;
    float cs = 0.f;
#pragma unroll
    for (int r = 0; r < 8; ++r) {
      const int row = r + hi * 8;
      cs += __expf(acc[t][r] - lse_s[row]);
    }
    const float other = __shfl_xor(cs, 16, 32);
    if (hi == 0) atomicAdd(&S[col], cs + other);
  }

  // quantized[b] = protos_f32[argmax], b128 copies
  for (int i = tid; i < 16 * (CDIM / 4); i += 256) {
    const int r = i >> 6, c4 = i & 63;
    reinterpret_cast<float4*>(out_q)[(size_t)(b0 + r) * (CDIM / 4) + c4] =
        reinterpret_cast<const float4*>(protos_f32)[(size_t)aidx_s[r] * (CDIM / 4) + c4];
  }
}

// ---------------------------------------------------------------------------
// k_loss: prior_k = S_k/B + EPS ; L_k = (T_k - sum lse)/B
// capacity = sum prior*(log prior - L_k) ; ent = -sum prior*log prior
__global__ __launch_bounds__(256) void k_loss(const float* __restrict__ S,
                                              const float* __restrict__ Tnd,
                                              const float* __restrict__ lse_sum,
                                              float* __restrict__ out_loss) {
  __shared__ float scap[256], sent[256];
  const int tid = threadIdx.x;
  const float lsum = *lse_sum;
  const float invB = 1.f / (float)BATCH;
  float cap = 0.f, ent = 0.f;
  for (int k = tid; k < KPROT; k += 256) {
    const float prior = S[k] * invB + 1e-6f;
    const float lp = __logf(prior);
    const float Lk = (Tnd[k] - lsum) * invB;
    cap += prior * (lp - Lk);
    ent -= prior * lp;
  }
  scap[tid] = cap; sent[tid] = ent;
  __syncthreads();
  for (int s = 128; s > 0; s >>= 1) {
    if (tid < s) { scap[tid] += scap[tid + s]; sent[tid] += sent[tid + s]; }
    __syncthreads();
  }
  if (tid == 0) out_loss[0] = 0.01f * scap[0] + (-0.001f) * sent[0];
}

// ---------------------------------------------------------------------------
extern "C" void kernel_launch(void* const* d_in, const int* in_sizes, int n_in,
                              void* d_out, int out_size, void* d_ws,
                              size_t ws_size, hipStream_t stream) {
  const float* x      = (const float*)d_in[0];
  const float* gumbel = (const float*)d_in[1];
  const float* W_emb  = (const float*)d_in[2];
  const float* b_emb  = (const float*)d_in[3];
  const float* W0     = (const float*)d_in[4];
  const float* b0     = (const float*)d_in[5];
  const float* W1     = (const float*)d_in[6];
  const float* b1     = (const float*)d_in[7];
  const float* W_mu   = (const float*)d_in[8];
  const float* b_mu   = (const float*)d_in[9];
  // d_in[10]=W_var, d_in[11]=b_var: logvar is dead code in the reference.
  const float* protos = (const float*)d_in[12];

  char* ws = (char*)d_ws;
  size_t off = 0;
  auto take = [&](size_t bytes) -> char* {
    off = (off + 255) & ~(size_t)255;
    char* p = ws + off;
    off += bytes;
    return p;
  };
  _Float16* tWemb  = (_Float16*)take((size_t)D_IN * HID * 2);
  _Float16* tW0    = (_Float16*)take((size_t)HID * HID * 2);
  _Float16* tW1    = (_Float16*)take((size_t)HID * CDIM * 2);
  _Float16* tWmu   = (_Float16*)take((size_t)CDIM * CDIM * 2);
  _Float16* prot_h = (_Float16*)take((size_t)KPROT * CDIM * 2);
  _Float16* mu_h   = (_Float16*)take((size_t)BATCH * CDIM * 2);
  float* mu_norm   = (float*)take((size_t)BATCH * 4);
  float* pnorm     = (float*)take((size_t)KPROT * 4);
  float* Sacc      = (float*)take((size_t)KPROT * 4);
  float* Tacc      = (float*)take((size_t)KPROT * 4);
  float* lse_sum   = (float*)take(16);

  float* out_q    = (float*)d_out;
  float* out_loss = out_q + (size_t)BATCH * CDIM;

  k_prep<<<512, 256, 0, stream>>>(W_emb, W0, W1, W_mu, protos, tWemb, tW0, tW1,
                                  tWmu, prot_h, pnorm, Sacc, Tacc, lse_sum);
  k_enc<<<BATCH / 16, 32, 0, stream>>>(x, tWemb, b_emb, tW0, b0, tW1, b1, tWmu,
                                       b_mu, mu_h, mu_norm);
  k_dist<<<BATCH / 16, 256, 0, stream>>>(mu_h, mu_norm, prot_h, pnorm, gumbel,
                                         protos, Sacc, Tacc, lse_sum, out_q);
  k_loss<<<1, 256, 0, stream>>>(Sacc, Tacc, lse_sum, out_loss);
}
